// AttnApproximator_55568286875985
// MI455X (gfx1250) — matrix operand order repaired and checked by reading, not compiled, not run on hardware
//
#include <hip/hip_runtime.h>

#define B_   2
#define S_   1024
#define H_   4096
#define NH_  32
#define HD_  128
#define M_   (B_ * S_)

typedef __attribute__((ext_vector_type(16))) __bf16 v16bf;
typedef __attribute__((ext_vector_type(8)))  float  v8f;

__device__ inline unsigned short f2bf(float f) {
    union { float f; unsigned u; } c; c.f = f;
    unsigned u = c.u;
    unsigned r = u + 0x7FFFu + ((u >> 16) & 1u);   // round-to-nearest-even
    return (unsigned short)(r >> 16);
}

__device__ inline v8f wmma_bf16(v16bf a, v16bf b, v8f c) {
    // D = A(16x32 bf16) * B(32x16 bf16) + C(16x16 f32)
    return __builtin_amdgcn_wmma_f32_16x16x32_bf16(
        false, a, false, b, (short)0, c, false, false);
}

union F16x16 { v16bf v; unsigned u[8]; };

// LDS byte address (wave-relative) from a generic pointer to __shared__:
// generic LDS addresses are {aperture_hi32, lds_offset32} -> take low 32 bits.
__device__ inline unsigned lds_addr(const void* p) {
    return (unsigned)(unsigned long long)p;
}

// ---------------------------------------------------------------------------
// f32 -> bf16 cast
// ---------------------------------------------------------------------------
__global__ void cast_bf16_kernel(const float* __restrict__ src,
                                 unsigned short* __restrict__ dst, int n) {
    int i = blockIdx.x * blockDim.x + threadIdx.x;
    if (i < n) dst[i] = f2bf(src[i]);
}

// ---------------------------------------------------------------------------
// RoPE cos/sin tables: [S][HD], duplicated halves
// ---------------------------------------------------------------------------
__global__ void rope_tables_kernel(float* __restrict__ cosT,
                                   float* __restrict__ sinT) {
    int i = blockIdx.x * blockDim.x + threadIdx.x;     // S * HD/2 threads
    if (i >= S_ * (HD_ / 2)) return;
    int d = i & (HD_ / 2 - 1);
    int s = i >> 6;                                    // HD/2 == 64
    float inv_freq = __expf(-((2.0f * d) / HD_) * __logf(10000.0f));
    float f = (float)s * inv_freq;
    float c = __cosf(f), sn = __sinf(f);
    cosT[s * HD_ + d] = c;  cosT[s * HD_ + d + HD_ / 2] = c;
    sinT[s * HD_ + d] = sn; sinT[s * HD_ + d + HD_ / 2] = sn;
}

// ---------------------------------------------------------------------------
// GEMM: C[M,N](f32) = A[M,K](bf16) @ W[N,K](bf16)^T  (torch Linear semantics)
// Block 256 thr = 8 waves, tile 128x128x32. Wave w -> rows [w*16, w*16+16).
// Global->LDS staging uses CDNA5 async-to-LDS copies (ASYNCcnt), double
// buffered so stage t+1's DMA overlaps stage t's WMMAs.
// ---------------------------------------------------------------------------
#define BK_  32
#define LDT_ 40   // padded LDS row stride (bf16 units) to spread banks

__global__ __launch_bounds__(256)
void gemm_bf16_nt(const unsigned short* __restrict__ A,
                  const unsigned short* __restrict__ W,
                  float* __restrict__ C, int M, int N, int K) {
    __shared__ unsigned short As[2][128 * LDT_];
    __shared__ unsigned short Ws[2][128 * LDT_];
    const int tid  = threadIdx.x;
    const int lane = tid & 31;
    const int wave = tid >> 5;
    const int bm = blockIdx.y * 128;
    const int bn = blockIdx.x * 128;
    const int mrow = lane & 15;
    const int ncol = lane & 15;
    const int hi   = lane >> 4;     // wave half: 0 or 1

    v8f acc[8];
#pragma unroll
    for (int t = 0; t < 8; ++t)
#pragma unroll
        for (int e = 0; e < 8; ++e) acc[t][e] = 0.0f;

    // Issue one stage's tile DMA: 512 16B chunks over 256 threads, 2 arrays
    // -> 4 async-load instructions per wave per stage (in-order completion).
    auto issue_stage = [&](int kk, int buf) {
#pragma unroll
        for (int i = 0; i < 2; ++i) {
            int chunk = tid + i * 256;
            int r = chunk >> 2;
            int c = (chunk & 3) * 8;            // 8 bf16 = 16 bytes
            unsigned lA = lds_addr(&As[buf][r * LDT_ + c]);
            unsigned lW = lds_addr(&Ws[buf][r * LDT_ + c]);
            const unsigned short* gA = &A[(size_t)(bm + r) * K + kk + c];
            const unsigned short* gW = &W[(size_t)(bn + r) * K + kk + c];
            asm volatile("global_load_async_to_lds_b128 %0, %1, off"
                         :: "v"(lA), "v"(gA) : "memory");
            asm volatile("global_load_async_to_lds_b128 %0, %1, off"
                         :: "v"(lW), "v"(gW) : "memory");
        }
    };

    issue_stage(0, 0);
    const int nsteps = K / BK_;
    for (int it = 0; it < nsteps; ++it) {
        const int cur = it & 1;
        if (it + 1 < nsteps) {
            issue_stage((it + 1) * BK_, cur ^ 1);
            // 8 async loads outstanding; oldest 4 (current stage) must land.
            asm volatile("s_wait_asynccnt 0x4" ::: "memory");
        } else {
            asm volatile("s_wait_asynccnt 0x0" ::: "memory");
        }
        __syncthreads();

        F16x16 af;
#pragma unroll
        for (int v = 0; v < 8; ++v) {           // A: contiguous k = 16*hi + 2v (this code's order)
            int kb = (hi ? 16 : 0) + 2 * v;
            af.u[v] = *(const unsigned*)&As[cur][(wave * 16 + mrow) * LDT_ + kb];
        }
#pragma unroll
        for (int t = 0; t < 8; ++t) {
            F16x16 bf;
#pragma unroll
            for (int v = 0; v < 8; ++v)         // B: same contiguous k order as A
                bf.u[v] = *(const unsigned*)&Ws[cur][(t * 16 + ncol) * LDT_ + (hi ? 16 : 0) + 2 * v];
            acc[t] = wmma_bf16(af.v, bf.v, acc[t]);
        }
        __syncthreads();                        // stage buffer reusable
    }

#pragma unroll
    for (int t = 0; t < 8; ++t)
#pragma unroll
        for (int r = 0; r < 8; ++r)             // C layout: vgpr r -> M=r(+8)
            C[(size_t)(bm + wave * 16 + r + hi * 8) * N + bn + t * 16 + ncol] = acc[t][r];
}

// ---------------------------------------------------------------------------
// RoPE + repack: Qf/Kf/Vf [B*S, H] f32 -> Qb/Kb [B,NH,S,HD] bf16,
//                Vt [B,NH,HD,S] bf16 (transposed for P.V WMMA B-frags)
// ---------------------------------------------------------------------------
__global__ void rope_pack_kernel(const float* __restrict__ Qf,
                                 const float* __restrict__ Kf,
                                 const float* __restrict__ Vf,
                                 const float* __restrict__ cosT,
                                 const float* __restrict__ sinT,
                                 const int* __restrict__ posids,
                                 unsigned short* __restrict__ Qb,
                                 unsigned short* __restrict__ Kb,
                                 unsigned short* __restrict__ Vt) {
    int idx = blockIdx.x * blockDim.x + threadIdx.x;   // B*NH*S*HD threads
    if (idx >= B_ * NH_ * S_ * HD_) return;
    int d = idx & (HD_ - 1);
    int s = (idx >> 7) & (S_ - 1);
    int h = (idx >> 17) & (NH_ - 1);
    int b = idx >> 22;

    int pos = posids[b * S_ + s];
    float c  = cosT[pos * HD_ + d];
    float sn = sinT[pos * HD_ + d];

    size_t row = (size_t)(b * S_ + s) * H_;
    int col  = h * HD_ + d;
    int dpar = (d < HD_ / 2) ? (d + HD_ / 2) : (d - HD_ / 2);
    float sgn = (d < HD_ / 2) ? -1.0f : 1.0f;

    float q  = Qf[row + col];
    float k  = Kf[row + col];
    float qr = sgn * Qf[row + h * HD_ + dpar];
    float kr = sgn * Kf[row + h * HD_ + dpar];

    size_t o = ((size_t)(b * NH_ + h) * S_ + s) * HD_ + d;
    Qb[o] = f2bf(q * c + qr * sn);
    Kb[o] = f2bf(k * c + kr * sn);
    Vt[((size_t)(b * NH_ + h) * HD_ + d) * S_ + s] = f2bf(Vf[row + col]);
}

// ---------------------------------------------------------------------------
// Flash attention, one wave per (b, h, 16-row q tile).
// scores = Q.K^T/sqrt(HD); Z2 over ALL kv; causal online softmax; ctx = P.V
// ---------------------------------------------------------------------------
__global__ __launch_bounds__(32)
void attn_kernel(const unsigned short* __restrict__ Qb,
                 const unsigned short* __restrict__ Kb,
                 const unsigned short* __restrict__ Vt,
                 unsigned short* __restrict__ Ctxb,
                 float* __restrict__ z2part) {
    const int b = blockIdx.z, h = blockIdx.y, qt = blockIdx.x;
    const int qbase = qt * 16;
    const int lane = threadIdx.x & 31;
    const int mrow = lane & 15;
    const int ncol = lane & 15;
    const int hi   = lane >> 4;

    const size_t head = (size_t)(b * NH_ + h);
    const unsigned short* Q  = Qb + head * S_ * HD_;
    const unsigned short* Kp = Kb + head * S_ * HD_;
    const unsigned short* Vp = Vt + head * HD_ * S_;   // [HD][S]

    __shared__ unsigned short Pst[16 * LDT_];          // P staging (C->A relayout)

    // Q fragments for all 4 K-steps of HD=128
    F16x16 qf[4];
#pragma unroll
    for (int kk = 0; kk < 4; ++kk)
#pragma unroll
        for (int v = 0; v < 8; ++v) {
            int kb = (hi ? 16 : 0) + 2 * v;
            qf[kk].u[v] = *(const unsigned*)&Q[(size_t)(qbase + mrow) * HD_ + kk * 32 + kb];
        }

    float mrun[8], lrun[8], z2a[8];
    v8f acc[8];
#pragma unroll
    for (int r = 0; r < 8; ++r) { mrun[r] = -__builtin_inff(); lrun[r] = 0.f; z2a[r] = 0.f; }
#pragma unroll
    for (int t = 0; t < 8; ++t)
#pragma unroll
        for (int e = 0; e < 8; ++e) acc[t][e] = 0.0f;

    const float sc = 0.08838834764831845f;             // 1/sqrt(128)

    for (int j0 = 0; j0 < S_; j0 += 32) {
        // ---- scores for two 16-wide kv sub-tiles
        v8f sf[2];
#pragma unroll
        for (int sub = 0; sub < 2; ++sub) {
            v8f s;
#pragma unroll
            for (int e = 0; e < 8; ++e) s[e] = 0.0f;
#pragma unroll
            for (int kk = 0; kk < 4; ++kk) {
                F16x16 bf;
#pragma unroll
                for (int v = 0; v < 8; ++v)
                    bf.u[v] = *(const unsigned*)&Kp[(size_t)(j0 + sub * 16 + ncol) * HD_ +
                                                    kk * 32 + (hi ? 16 : 0) + 2 * v];
                s = wmma_bf16(qf[kk].v, bf.v, s);
            }
#pragma unroll
            for (int r = 0; r < 8; ++r) {
                s[r] *= sc;
                z2a[r] += __expf(s[r]);                // Z2: UNMASKED exp-sum
            }
            sf[sub] = s;
        }

        // ---- causal softmax / ctx update (skip fully-masked tiles)
        if (j0 <= qbase + 15) {
            float pm0[8], pm1[8], rowm[8], rows[8];
#pragma unroll
            for (int r = 0; r < 8; ++r) {
                int qg = qbase + r + hi * 8;
                pm0[r] = (j0 + ncol       <= qg) ? sf[0][r] : -__builtin_inff();
                pm1[r] = (j0 + 16 + ncol  <= qg) ? sf[1][r] : -__builtin_inff();
                rowm[r] = fmaxf(pm0[r], pm1[r]);
            }
#pragma unroll
            for (int o = 1; o <= 8; o <<= 1)
#pragma unroll
                for (int r = 0; r < 8; ++r)
                    rowm[r] = fmaxf(rowm[r], __shfl_xor(rowm[r], o, 32));

            float scalef[8];
#pragma unroll
            for (int r = 0; r < 8; ++r) {
                float mn = fmaxf(mrun[r], rowm[r]);
                scalef[r] = __expf(mrun[r] - mn);
                mrun[r] = mn;
                float p0 = __expf(pm0[r] - mn);
                float p1 = __expf(pm1[r] - mn);
                rows[r] = p0 + p1;
                int Mr = r + hi * 8;
                Pst[Mr * LDT_ + ncol]      = f2bf(p0);
                Pst[Mr * LDT_ + 16 + ncol] = f2bf(p1);
            }
#pragma unroll
            for (int o = 1; o <= 8; o <<= 1)
#pragma unroll
                for (int r = 0; r < 8; ++r)
                    rows[r] += __shfl_xor(rows[r], o, 32);
#pragma unroll
            for (int r = 0; r < 8; ++r)
                lrun[r] = lrun[r] * scalef[r] + rows[r];
#pragma unroll
            for (int t = 0; t < 8; ++t)
#pragma unroll
                for (int r = 0; r < 8; ++r) acc[t][r] *= scalef[r];

            __syncthreads();
            F16x16 pf;                                  // P as A-fragment 16x32
#pragma unroll
            for (int v = 0; v < 8; ++v) {
                int kb = (hi ? 16 : 0) + 2 * v;
                pf.u[v] = *(const unsigned*)&Pst[mrow * LDT_ + kb];
            }
#pragma unroll
            for (int t = 0; t < 8; ++t) {               // ctx += P.V
                F16x16 bf;
#pragma unroll
                for (int v = 0; v < 8; ++v)
                    bf.u[v] = *(const unsigned*)&Vp[(size_t)(t * 16 + ncol) * S_ +
                                                    j0 + (hi ? 16 : 0) + 2 * v];
                acc[t] = wmma_bf16(pf.v, bf.v, acc[t]);
            }
            __syncthreads();
        }
    }

    // ---- normalize + write ctx (bf16, [B,S, NH*HD] row-major)
    float invl[8];
#pragma unroll
    for (int r = 0; r < 8; ++r) invl[r] = 1.0f / lrun[r];
#pragma unroll
    for (int t = 0; t < 8; ++t)
#pragma unroll
        for (int r = 0; r < 8; ++r) {
            size_t row = (size_t)(b * S_ + qbase + r + hi * 8);
            Ctxb[row * H_ + h * HD_ + t * 16 + ncol] = f2bf(acc[t][r] * invl[r]);
        }

    // ---- Z2 per-head partial: reduce across the 16-lane group, write once
#pragma unroll
    for (int o = 1; o <= 8; o <<= 1)
#pragma unroll
        for (int r = 0; r < 8; ++r) z2a[r] += __shfl_xor(z2a[r], o, 32);
    if (ncol == 0) {
#pragma unroll
        for (int r = 0; r < 8; ++r)
            z2part[head * S_ + qbase + r + hi * 8] = z2a[r];
    }
}

// ---------------------------------------------------------------------------
// Z2[b,s] = mean over heads of z2part
// ---------------------------------------------------------------------------
__global__ void z2_reduce_kernel(const float* __restrict__ z2part,
                                 float* __restrict__ Z2) {
    int i = blockIdx.x * blockDim.x + threadIdx.x;     // B*S threads
    if (i >= B_ * S_) return;
    int b = i / S_, s = i % S_;
    float acc = 0.f;
#pragma unroll
    for (int h = 0; h < NH_; ++h)
        acc += z2part[(size_t)(b * NH_ + h) * S_ + s];
    Z2[i] = acc * (1.0f / NH_);
}

// ---------------------------------------------------------------------------
extern "C" void kernel_launch(void* const* d_in, const int* in_sizes, int n_in,
                              void* d_out, int out_size, void* d_ws, size_t ws_size,
                              hipStream_t stream) {
    const float* hs = (const float*)d_in[0];
    const float* Wq = (const float*)d_in[1];
    const float* Wk = (const float*)d_in[2];
    const float* Wv = (const float*)d_in[3];
    const float* Wo = (const float*)d_in[4];
    const int*  pos = (const int*)d_in[5];
    float* out = (float*)d_out;                         // [B,S,H] f32
    float* Z2  = out + (size_t)M_ * H_;                 // [B,S]   f32

    char* ws = (char*)d_ws;
    size_t off = 0;
    auto alloc = [&](size_t bytes) -> char* {
        char* p = ws + off;
        off = (off + bytes + 255) & ~(size_t)255;
        return p;
    };

    float* cosT = (float*)alloc((size_t)S_ * HD_ * 4);
    float* sinT = (float*)alloc((size_t)S_ * HD_ * 4);
    unsigned short* Xb  = (unsigned short*)alloc((size_t)M_ * H_ * 2);
    unsigned short* Wqb = (unsigned short*)alloc((size_t)H_ * H_ * 2);
    unsigned short* Wkb = (unsigned short*)alloc((size_t)H_ * H_ * 2);
    unsigned short* Wvb = (unsigned short*)alloc((size_t)H_ * H_ * 2);
    unsigned short* Wob = (unsigned short*)alloc((size_t)H_ * H_ * 2);
    float* Qf = (float*)alloc((size_t)M_ * H_ * 4);
    float* Kf = (float*)alloc((size_t)M_ * H_ * 4);
    float* Vf = (float*)alloc((size_t)M_ * H_ * 4);
    unsigned short* Qbh = (unsigned short*)alloc((size_t)B_ * NH_ * S_ * HD_ * 2);
    unsigned short* Kbh = (unsigned short*)alloc((size_t)B_ * NH_ * S_ * HD_ * 2);
    unsigned short* Vth = (unsigned short*)alloc((size_t)B_ * NH_ * S_ * HD_ * 2);
    unsigned short* Ctxb = (unsigned short*)alloc((size_t)M_ * H_ * 2);
    float* z2part = (float*)alloc((size_t)B_ * NH_ * S_ * 4);

    // 1) casts to bf16
    {
        int nX = M_ * H_;
        cast_bf16_kernel<<<(nX + 255) / 256, 256, 0, stream>>>(hs, Xb, nX);
        int nW = H_ * H_;
        cast_bf16_kernel<<<(nW + 255) / 256, 256, 0, stream>>>(Wq, Wqb, nW);
        cast_bf16_kernel<<<(nW + 255) / 256, 256, 0, stream>>>(Wk, Wkb, nW);
        cast_bf16_kernel<<<(nW + 255) / 256, 256, 0, stream>>>(Wv, Wvb, nW);
        cast_bf16_kernel<<<(nW + 255) / 256, 256, 0, stream>>>(Wo, Wob, nW);
    }
    // 2) RoPE tables
    {
        int n = S_ * (HD_ / 2);
        rope_tables_kernel<<<(n + 255) / 256, 256, 0, stream>>>(cosT, sinT);
    }
    // 3) Q/K/V projections (x @ W^T) via bf16 WMMA GEMM
    {
        dim3 grid(H_ / 128, M_ / 128);
        gemm_bf16_nt<<<grid, 256, 0, stream>>>(Xb, Wqb, Qf, M_, H_, H_);
        gemm_bf16_nt<<<grid, 256, 0, stream>>>(Xb, Wkb, Kf, M_, H_, H_);
        gemm_bf16_nt<<<grid, 256, 0, stream>>>(Xb, Wvb, Vf, M_, H_, H_);
    }
    // 4) RoPE + pack into per-head layouts
    {
        int n = B_ * NH_ * S_ * HD_;
        rope_pack_kernel<<<(n + 255) / 256, 256, 0, stream>>>(
            Qf, Kf, Vf, cosT, sinT, pos, Qbh, Kbh, Vth);
    }
    // 5) flash attention + Z2 partials
    {
        dim3 grid(S_ / 16, NH_, B_);
        attn_kernel<<<grid, 32, 0, stream>>>(Qbh, Kbh, Vth, Ctxb, z2part);
    }
    // 6) output projection -> d_out
    {
        dim3 grid(H_ / 128, M_ / 128);
        gemm_bf16_nt<<<grid, 256, 0, stream>>>(Ctxb, Wob, out, M_, H_, H_);
    }
    // 7) Z2 mean over heads
    {
        int n = B_ * S_;
        z2_reduce_kernel<<<(n + 255) / 256, 256, 0, stream>>>(z2part, Z2);
    }
}
